// AutoregressiveDecoder_88304527605901
// MI455X (gfx1250) — matrix operand order repaired and checked by reading, not compiled, not run on hardware
//
#include <hip/hip_runtime.h>
#include <hip/hip_bf16.h>
#include <math.h>

typedef __attribute__((ext_vector_type(16))) _Float16 v16h;
typedef __attribute__((ext_vector_type(4)))  _Float16 v4h;
typedef __attribute__((ext_vector_type(8)))  float    v8f;

#define FC 80
#define PRE_H 256
#define ENC 512
#define LSTMH 1024
#define ATT 128
#define LOCF 32
#define LOCK 31
#define TTOK 128
#define BATCH 32
#define TFRM 512

#define AS_STRIDE 68   // halves; 136B row pitch: 8B-aligned b64 stores, bank-spread

// epilogue flags
#define FLG_ACC   1
#define FLG_RELU  (1 << 1)
#define FLG_TANH  (2 << 1)

__device__ __forceinline__ float fast_sig(float x)
{
    return 1.0f / (1.0f + __expf(-x));
}
__device__ __forceinline__ float fast_tanh(float x)
{
    x = fminf(fmaxf(x, -15.0f), 15.0f);
    float e = __expf(2.0f * x);
    return (e - 1.0f) / (e + 1.0f);
}

// row remap: identity + offset, or padded PostNet conv layout
// (batch pitch 516 rows of 512 time steps): row = (m>>9)*516 + (m&511) + off
__device__ __forceinline__ int rowmap(int m, int remap, int off)
{
    return remap ? (((m >> 9) * (TFRM + 4)) + (m & (TFRM - 1)) + off) : (m + off);
}

__device__ __forceinline__ v16h cvtB(const float4* qv)
{
    v16h bf;
    bf[0]  = (_Float16)qv[0].x; bf[1]  = (_Float16)qv[0].y;
    bf[2]  = (_Float16)qv[0].z; bf[3]  = (_Float16)qv[0].w;
    bf[4]  = (_Float16)qv[1].x; bf[5]  = (_Float16)qv[1].y;
    bf[6]  = (_Float16)qv[1].z; bf[7]  = (_Float16)qv[1].w;
    bf[8]  = (_Float16)qv[2].x; bf[9]  = (_Float16)qv[2].y;
    bf[10] = (_Float16)qv[2].z; bf[11] = (_Float16)qv[2].w;
    bf[12] = (_Float16)qv[3].x; bf[13] = (_Float16)qv[3].y;
    bf[14] = (_Float16)qv[3].z; bf[15] = (_Float16)qv[3].w;
    return bf;
}

// ---------------------------------------------------------------------------
// Generic WMMA GEMM:  C[m,n] = act( (acc? C : 0) + A_row(m) . W_row(n) )
// A element (m,k): A[rowmap(m,aRemap,aOff)*lda + k]   (rowmap<0 reads as zero)
// W element (n,k): W[n*ldw + k*WSTRIDE + wOff]        (WSTRIDE=5 slices conv taps)
// C element (m,n): C[rowmap(m,cRemap,cOff)*ldc + n]
// 8 waves/block share an LDS-staged 16x64 f16 A tile; each wave owns one 16x16
// output tile and issues two v_wmma_f32_16x16x32_f16 per 64-wide K chunk.
// Both A and B global loads are software-pipelined one chunk ahead so their
// latency hides behind the barriers + LDS reads + WMMAs of the current chunk.
// Requires K % 16 == 0 (true for every GEMM in this network).
// ---------------------------------------------------------------------------
template <int WSTRIDE>
__global__ __launch_bounds__(256) void k_wmma_gemm(
    const float* __restrict__ A, int lda, int aRemap, int aOff,
    const float* __restrict__ W, int ldw, int wOff,
    float* __restrict__ C, int ldc, int cRemap, int cOff,
    const float* __restrict__ bias, const float* __restrict__ scale, float scaleMul,
    int M, int N, int K, int flags)
{
    __shared__ _Float16 As[16 * AS_STRIDE];
    const int tid   = threadIdx.x;
    const int wave  = tid >> 5;
    const int lane  = tid & 31;
    const int hl    = lane >> 4;       // 16-lane half
    const int l16   = lane & 15;
    const int tileM = blockIdx.y << 4;
    const int tileN = (blockIdx.x << 7) + (wave << 4);
    const int n     = tileN + l16;
    const bool nok  = (n < N);

    // each thread stages one contiguous 4-float run of the 16x64 A tile
    const int arow  = tid >> 4;          // 0..15
    const int acol  = (tid & 15) << 2;   // 0,4,...,60
    const int am    = tileM + arow;
    const int amrow = rowmap(am, aRemap, aOff);
    const bool arok = (am < M) && (amrow >= 0);
    const float* arp = A + (size_t)(arok ? amrow : 0) * lda;

    // loop-invariant W row base (clamped when n >= N; loads are guarded anyway)
    const float* wp = W + (size_t)(nok ? n : 0) * ldw + wOff;

    auto loadA = [&](int k0) -> float4 {
        float4 v = make_float4(0.0f, 0.0f, 0.0f, 0.0f);
        int gk = k0 + acol;
        if (arok && (gk + 4) <= K)
            v = *reinterpret_cast<const float4*>(arp + gk);
        return v;
    };
    auto loadBraw = [&](int gkbase, float4* qv) {
        if (nok && (gkbase + 16) <= K) {
            if (WSTRIDE == 1) {
                const float4* p = reinterpret_cast<const float4*>(wp + gkbase);
                qv[0] = p[0]; qv[1] = p[1]; qv[2] = p[2]; qv[3] = p[3];
            } else {
#pragma unroll
                for (int j = 0; j < 4; ++j) {
                    qv[j].x = wp[(size_t)(gkbase + 4 * j + 0) * WSTRIDE];
                    qv[j].y = wp[(size_t)(gkbase + 4 * j + 1) * WSTRIDE];
                    qv[j].z = wp[(size_t)(gkbase + 4 * j + 2) * WSTRIDE];
                    qv[j].w = wp[(size_t)(gkbase + 4 * j + 3) * WSTRIDE];
                }
            }
        } else {
            float4 z = make_float4(0.0f, 0.0f, 0.0f, 0.0f);
            qv[0] = z; qv[1] = z; qv[2] = z; qv[3] = z;
        }
    };

    // prologue: prefetch chunk 0
    float4 aP = loadA(0);
    float4 b0P[4], b1P[4];
    loadBraw(16 * hl, b0P);
    loadBraw(32 + 16 * hl, b1P);

    v8f acc = {};
    for (int k0 = 0; k0 < K; k0 += 64) {
        const bool more = (k0 + 64) < K;
        __syncthreads();
        {   // commit prefetched A run to LDS, then issue next chunk's A load
            v4h hv;
            hv[0] = (_Float16)aP.x; hv[1] = (_Float16)aP.y;
            hv[2] = (_Float16)aP.z; hv[3] = (_Float16)aP.w;
            *reinterpret_cast<v4h*>(&As[arow * AS_STRIDE + acol]) = hv;
        }
        if (more) aP = loadA(k0 + 64);
        __syncthreads();

        // convert prefetched B, then issue next chunk's B loads
        v16h bf0 = cvtB(b0P);
        v16h bf1 = cvtB(b1P);
        if (more) {
            loadBraw(k0 + 64 + 16 * hl, b0P);
            loadBraw(k0 + 96 + 16 * hl, b1P);
        }

        // A fragments for the two 32-wide sub-chunks (contiguous k = 16*half + i, matching B)
        v16h af0, af1;
#pragma unroll
        for (int i = 0; i < 16; ++i) {
            int kl = 16 * hl + i;
            af0[i] = As[l16 * AS_STRIDE + kl];
            af1[i] = As[l16 * AS_STRIDE + 32 + kl];
        }

        acc = __builtin_amdgcn_wmma_f32_16x16x32_f16(
            false, af0, false, bf0, (short)0, acc, false, false);
        acc = __builtin_amdgcn_wmma_f32_16x16x32_f16(
            false, af1, false, bf1, (short)0, acc, false, false);
    }

    // epilogue: C/D layout row = r + 8*half, col = lane&15
#pragma unroll
    for (int r = 0; r < 8; ++r) {
        int m = tileM + r + 8 * hl;
        if (m < M && nok) {
            size_t cidx = (size_t)rowmap(m, cRemap, cOff) * ldc + n;
            float v = acc[r];
            if (flags & FLG_ACC) v += C[cidx];
            if (scale) v *= scale[n] * scaleMul;
            if (bias)  v += bias[n];
            int act = (flags >> 1) & 3;
            if (act == 1)      v = fmaxf(v, 0.0f);
            else if (act == 2) v = fast_tanh(v);
            C[cidx] = v;
        }
    }
}

// ---------------------------------------------------------------------------
// LSTM gating: gates [B, 4*1024] in PyTorch order i,f,g,o
// ---------------------------------------------------------------------------
__global__ void k_lstm_gate(const float* __restrict__ gates,
                            float* __restrict__ h, float* __restrict__ c)
{
    int idx = blockIdx.x * blockDim.x + threadIdx.x;
    if (idx >= BATCH * LSTMH) return;
    int b = idx >> 10, u = idx & 1023;
    const float* g = gates + (size_t)b * 4096;
    float ig = fast_sig(g[u]);
    float fg = fast_sig(g[1024 + u]);
    float gg = fast_tanh(g[2048 + u]);
    float og = fast_sig(g[3072 + u]);
    float cn = fg * c[idx] + ig * gg;
    c[idx] = cn;
    h[idx] = og * fast_tanh(cn);
}

// ---------------------------------------------------------------------------
// Location-sensitive attention for one step. block = batch row b, thread = token t.
// ---------------------------------------------------------------------------
__global__ void k_attention(const float* __restrict__ tok,     // [TTOK][B][ATT]
                            const float* __restrict__ q,       // [B][ATT]
                            const float* __restrict__ loc_conv,// [32][1][31]
                            const float* __restrict__ loc_w,   // [ATT][32]
                            const float* __restrict__ attn_v,  // [ATT]
                            float* __restrict__ align_prev,    // [B][TTOK]
                            float* __restrict__ ctx,           // [B][ATT]
                            float* __restrict__ align_out)     // d_out slice [B][TTOK]
{
    __shared__ float sA[TTOK];
    __shared__ float sR[TTOK];
    int b = blockIdx.x;
    int t = threadIdx.x;

    sA[t] = align_prev[b * TTOK + t];
    __syncthreads();

    // 1->32 channel conv over token axis, kernel 31, pad 15
    float loc[LOCF];
#pragma unroll
    for (int f = 0; f < LOCF; ++f) loc[f] = 0.0f;
    for (int k = 0; k < LOCK; ++k) {
        int p = t + k - 15;
        if (p >= 0 && p < TTOK) {
            float av = sA[p];
#pragma unroll
            for (int f = 0; f < LOCF; ++f) loc[f] += loc_conv[f * LOCK + k] * av;
        }
    }
    // energy e_t = sum_h v[h] * tanh(tok + q + loc_w.loc)
    float e = 0.0f;
    for (int h = 0; h < ATT; ++h) {
        float s = tok[((size_t)t * BATCH + b) * ATT + h] + q[b * ATT + h];
        float lp = 0.0f;
#pragma unroll
        for (int f = 0; f < LOCF; ++f) lp += loc_w[h * LOCF + f] * loc[f];
        e += attn_v[h] * fast_tanh(s + lp);
    }
    // softmax over 128 tokens
    sR[t] = e; __syncthreads();
    for (int s = 64; s > 0; s >>= 1) { if (t < s) sR[t] = fmaxf(sR[t], sR[t + s]); __syncthreads(); }
    float mx = sR[0]; __syncthreads();
    float ex = __expf(e - mx);
    sR[t] = ex; __syncthreads();
    for (int s = 64; s > 0; s >>= 1) { if (t < s) sR[t] += sR[t + s]; __syncthreads(); }
    float a = ex / sR[0];
    __syncthreads();
    sA[t] = a;
    align_prev[b * TTOK + t] = a;
    align_out[b * TTOK + t] = a;
    __syncthreads();
    // context: thread t doubles as attention-dim index (ATT == TTOK == 128)
    float cv = 0.0f;
    for (int j = 0; j < TTOK; ++j)
        cv += sA[j] * tok[((size_t)j * BATCH + b) * ATT + t];
    ctx[b * ATT + t] = cv;
}

__global__ void k_stop(const float* __restrict__ h2, const float* __restrict__ ctx,
                       const float* __restrict__ sw, const float* __restrict__ sb,
                       float* __restrict__ out)
{
    int b = threadIdx.x;
    if (b >= BATCH) return;
    float s = sb[0];
    for (int k = 0; k < LSTMH; ++k) s += h2[b * LSTMH + k] * sw[k];
    for (int k = 0; k < ATT; ++k)   s += ctx[b * ATT + k] * sw[LSTMH + k];
    out[b] = fast_sig(s);
}

__global__ void k_zero(float* __restrict__ p, size_t n)
{
    size_t i  = (size_t)blockIdx.x * 256 + threadIdx.x;
    size_t st = (size_t)gridDim.x * 256;
    for (; i < n; i += st) p[i] = 0.0f;
}

// pack frames [T][B][80] -> padded conv buffer [B][T+4][512-stride]
__global__ void k_pack(const float* __restrict__ frames, float* __restrict__ buf)
{
    int idx = blockIdx.x * 256 + threadIdx.x;
    if (idx >= TFRM * BATCH * FC) return;
    int t = idx / (BATCH * FC);
    int r = idx % (BATCH * FC);
    int b = r / FC, c = r % FC;
    buf[((size_t)b * (TFRM + 4) + t + 2) * 512 + c] = frames[idx];
}

__global__ void k_residual(const float* __restrict__ frames, const float* __restrict__ buf,
                           float* __restrict__ out)
{
    int idx = blockIdx.x * 256 + threadIdx.x;
    if (idx >= TFRM * BATCH * FC) return;
    int t = idx / (BATCH * FC);
    int r = idx % (BATCH * FC);
    int b = r / FC, c = r % FC;
    out[idx] = frames[idx] + buf[((size_t)b * (TFRM + 4) + t + 2) * 512 + c];
}

// ---------------------------------------------------------------------------
extern "C" void kernel_launch(void* const* d_in, const int* in_sizes, int n_in,
                              void* d_out, int out_size, void* d_ws, size_t ws_size,
                              hipStream_t stream)
{
    (void)in_sizes; (void)n_in; (void)out_size; (void)ws_size;

    const float* enc     = (const float*)d_in[0];
    const float* gtf     = (const float*)d_in[1];
    const float* pre_w1  = (const float*)d_in[2];
    const float* pre_b1  = (const float*)d_in[3];
    const float* pre_w2  = (const float*)d_in[4];
    const float* pre_b2  = (const float*)d_in[5];
    const float* l1_wih  = (const float*)d_in[6];
    const float* l1_whh  = (const float*)d_in[7];
    const float* l1_b    = (const float*)d_in[8];
    const float* l2_wih  = (const float*)d_in[9];
    const float* l2_whh  = (const float*)d_in[10];
    const float* l2_b    = (const float*)d_in[11];
    const float* tok_w   = (const float*)d_in[12];
    const float* tok_b   = (const float*)d_in[13];
    const float* attn_wq = (const float*)d_in[14];
    const float* attn_bq = (const float*)d_in[15];
    const float* loc_cv  = (const float*)d_in[16];
    const float* loc_w   = (const float*)d_in[17];
    const float* attn_v  = (const float*)d_in[18];
    const float* out_w   = (const float*)d_in[19];
    const float* out_b   = (const float*)d_in[20];
    const float* stop_w  = (const float*)d_in[21];
    const float* stop_b  = (const float*)d_in[22];
    const float* post_w[5]; const float* post_g[5]; const float* post_b[5];
    for (int i = 0; i < 5; ++i) {
        post_w[i] = (const float*)d_in[23 + i];
        post_g[i] = (const float*)d_in[28 + i];
        post_b[i] = (const float*)d_in[33 + i];
    }

    float* out = (float*)d_out;
    float* ws  = (float*)d_ws;

    // d_out sections (floats)
    const size_t OFF_FRAMES = 0;
    const size_t OFF_RES    = (size_t)TFRM * BATCH * FC;          // 1310720
    const size_t OFF_STOP   = OFF_RES * 2;                         // 2621440
    const size_t OFF_ALIGN  = OFF_STOP + (size_t)TFRM * BATCH;     // 2637824

    // workspace layout (floats). PostNet ping-pong buffers overlap the decode
    // scratch (dead after the scan).
    const size_t OFF_PRETMP = 0;                                   // 16384*256
    const size_t OFF_PRE    = 4194304;                             // 16384*256
    const size_t OFF_TOK    = 8388608;                             // 4096*128
    const size_t OFF_GATES  = 8912896;                             // 32*4096
    const size_t OFF_H1     = 9043968;
    const size_t OFF_C1     = 9076736;
    const size_t OFF_H2     = 9109504;
    const size_t OFF_C2     = 9142272;
    const size_t OFF_Q      = 9175040;
    const size_t OFF_CTX    = 9179136;
    const size_t OFF_APREV  = 9183232;                             // end 9187328
    const size_t OFF_BUF0   = 0;                                   // 32*516*512
    const size_t OFF_BUF1   = 8454144;                             // end 16908288

    float* pre_tmp = ws + OFF_PRETMP;
    float* pre     = ws + OFF_PRE;
    float* tok     = ws + OFF_TOK;
    float* gates   = ws + OFF_GATES;
    float* h1      = ws + OFF_H1;
    float* c1      = ws + OFF_C1;
    float* h2      = ws + OFF_H2;
    float* c2      = ws + OFF_C2;
    float* q       = ws + OFF_Q;
    float* ctx     = ws + OFF_CTX;
    float* aprev   = ws + OFF_APREV;

    auto gemm = [&](const float* A, int lda, int aRemap, int aOff,
                    const float* W, int ldw, int wStride, int wOff,
                    float* C, int ldc, int cRemap, int cOff,
                    const float* bias, const float* scale, float sm,
                    int M, int N, int K, int flags) {
        dim3 g((unsigned)((N + 127) / 128), (unsigned)((M + 15) / 16));
        if (wStride == 1)
            hipLaunchKernelGGL(k_wmma_gemm<1>, g, dim3(256), 0, stream,
                               A, lda, aRemap, aOff, W, ldw, wOff,
                               C, ldc, cRemap, cOff, bias, scale, sm, M, N, K, flags);
        else
            hipLaunchKernelGGL(k_wmma_gemm<5>, g, dim3(256), 0, stream,
                               A, lda, aRemap, aOff, W, ldw, wOff,
                               C, ldc, cRemap, cOff, bias, scale, sm, M, N, K, flags);
    };

    const int MT = TFRM * BATCH;     // 16384 prenet rows
    const int MK = TTOK * BATCH;     // 4096 token rows

    // ---- PreNet (teacher-forcing shift via rowOff = -BATCH) ----
    gemm(gtf, FC, 0, -BATCH, pre_w1, FC, 1, 0,
         pre_tmp, PRE_H, 0, 0, pre_b1, nullptr, 1.0f, MT, PRE_H, FC, FLG_RELU);
    gemm(pre_tmp, PRE_H, 0, 0, pre_w2, PRE_H, 1, 0,
         pre, PRE_H, 0, 0, pre_b2, nullptr, 1.0f, MT, PRE_H, PRE_H, FLG_RELU);

    // ---- token projection: tok[t][b][h] ----
    gemm(enc, ENC, 0, 0, tok_w, ENC, 1, 0,
         tok, ATT, 0, 0, tok_b, nullptr, 1.0f, MK, ATT, ENC, 0);

    // ---- zero recurrent state (h1,c1,h2,c2,q,ctx,align_prev are contiguous) ----
    hipLaunchKernelGGL(k_zero, dim3(128), dim3(256), 0, stream, h1, (size_t)143360);

    // ---- sequential decoder scan ----
    for (int t = 0; t < TFRM; ++t) {
        const float* pre_t = pre + (size_t)t * BATCH * PRE_H;
        float* frame_t = out + OFF_FRAMES + (size_t)t * BATCH * FC;

        // LSTM1 gates = [pre_t|ctx] @ wih^T + h1 @ whh^T + b
        gemm(pre_t, PRE_H, 0, 0, l1_wih, 384, 1, 0,
             gates, 4096, 0, 0, l1_b, nullptr, 1.0f, BATCH, 4096, PRE_H, 0);
        gemm(ctx, ATT, 0, 0, l1_wih, 384, 1, PRE_H,
             gates, 4096, 0, 0, nullptr, nullptr, 1.0f, BATCH, 4096, ATT, FLG_ACC);
        gemm(h1, LSTMH, 0, 0, l1_whh, LSTMH, 1, 0,
             gates, 4096, 0, 0, nullptr, nullptr, 1.0f, BATCH, 4096, LSTMH, FLG_ACC);
        hipLaunchKernelGGL(k_lstm_gate, dim3(128), dim3(256), 0, stream, gates, h1, c1);

        // attention query + location-sensitive attention
        gemm(h1, LSTMH, 0, 0, attn_wq, LSTMH, 1, 0,
             q, ATT, 0, 0, attn_bq, nullptr, 1.0f, BATCH, ATT, LSTMH, 0);
        hipLaunchKernelGGL(k_attention, dim3(BATCH), dim3(TTOK), 0, stream,
                           tok, q, loc_cv, loc_w, attn_v, aprev, ctx,
                           out + OFF_ALIGN + (size_t)t * BATCH * TTOK);

        // LSTM2 gates = [h1|ctx] @ wih^T + h2 @ whh^T + b
        gemm(h1, LSTMH, 0, 0, l2_wih, 1152, 1, 0,
             gates, 4096, 0, 0, l2_b, nullptr, 1.0f, BATCH, 4096, LSTMH, 0);
        gemm(ctx, ATT, 0, 0, l2_wih, 1152, 1, LSTMH,
             gates, 4096, 0, 0, nullptr, nullptr, 1.0f, BATCH, 4096, ATT, FLG_ACC);
        gemm(h2, LSTMH, 0, 0, l2_whh, LSTMH, 1, 0,
             gates, 4096, 0, 0, nullptr, nullptr, 1.0f, BATCH, 4096, LSTMH, FLG_ACC);
        hipLaunchKernelGGL(k_lstm_gate, dim3(128), dim3(256), 0, stream, gates, h2, c2);

        // frame = [h2|ctx] @ out_w^T + out_b ; stop token
        gemm(h2, LSTMH, 0, 0, out_w, 1152, 1, 0,
             frame_t, FC, 0, 0, out_b, nullptr, 1.0f, BATCH, FC, LSTMH, 0);
        gemm(ctx, ATT, 0, 0, out_w, 1152, 1, LSTMH,
             frame_t, FC, 0, 0, nullptr, nullptr, 1.0f, BATCH, FC, ATT, FLG_ACC);
        hipLaunchKernelGGL(k_stop, dim3(1), dim3(32), 0, stream,
                           h2, ctx, stop_w, stop_b, out + OFF_STOP + (size_t)t * BATCH);
    }

    // ---- PostNet: 5 conv layers as 5 accumulated GEMM taps each ----
    hipLaunchKernelGGL(k_zero, dim3(2048), dim3(256), 0, stream, ws, (size_t)16908288);
    hipLaunchKernelGGL(k_pack, dim3((TFRM * BATCH * FC + 255) / 256), dim3(256), 0, stream,
                       out + OFF_FRAMES, ws + OFF_BUF0);

    const int cin[5]  = {FC, 512, 512, 512, 512};
    const int cout[5] = {512, 512, 512, 512, FC};
    float* bufs[2] = {ws + OFF_BUF0, ws + OFF_BUF1};
    const float bninv = 1.0f / sqrtf(1.0f + 1e-5f);
    for (int L = 0; L < 5; ++L) {
        float* bin = bufs[L & 1];
        float* bo  = bufs[(L + 1) & 1];
        for (int k = 0; k < 5; ++k) {
            int flags = (k ? FLG_ACC : 0);
            if (k == 4 && L < 4) flags |= FLG_TANH;
            gemm(bin, 512, 1, k,
                 post_w[L], cin[L] * 5, 5, k,
                 bo, 512, 1, 2,
                 (k == 4 ? post_b[L] : nullptr), (k == 4 ? post_g[L] : nullptr), bninv,
                 MT, cout[L], cin[L], flags);
        }
    }
    hipLaunchKernelGGL(k_residual, dim3((TFRM * BATCH * FC + 255) / 256), dim3(256), 0, stream,
                       out + OFF_FRAMES, bufs[1], out + OFF_RES);
}
